// HRM_36747740184857
// MI455X (gfx1250) — matrix operand order repaired and checked by reading, not compiled, not run on hardware
//
#include <hip/hip_runtime.h>
#include <hip/hip_bf16.h>
#include <math.h>

// Problem constants (fixed by the reference).
constexpr int C_NODES = 262144;
constexpr int B_GRAPH = 4096;
constexpr int D_DIM   = 256;
constexpr int H_HEADS = 4;
constexpr int HD_DIM  = 64;

typedef __attribute__((ext_vector_type(16))) __bf16        v16bf;
typedef __attribute__((ext_vector_type(8)))  float         v8f;
typedef __attribute__((ext_vector_type(8)))  unsigned int  v8u;

// ---------------------------------------------------------------------------
// WMMA helpers (wave32, v_wmma_f32_16x16x32_bf16).
//
// A-matrix 16x32 bf16 (ISA 7.12.2): lane L -> m = L&15, half = L>>4.
//   VGPR v holds K pair k0 = half*16 + v*2 as loaded here (contiguous K, not the ISA table order).
// B-matrix 32x16 bf16: lane L -> n = L&15, half = L>>4.
//   VGPR v holds K pair k0 = half*16 + v*2 as loaded here (same contiguous order as A).
// C/D f32 16x16: lane L -> n = L&15; VGPR r -> m = r + 8*(L>>4).
// ---------------------------------------------------------------------------

__device__ __forceinline__ v8f wmma_bf16(v16bf a, v16bf b, v8f c) {
  return __builtin_amdgcn_wmma_f32_16x16x32_bf16(
      /*neg_a=*/false, a, /*neg_b=*/false, b,
      /*c_mod=*/(short)0, c, /*reuse_a=*/false, /*reuse_b=*/false);
}

// Load 16x32 A tile (f32 source, converted to bf16). `tile` points at [0][0].
__device__ __forceinline__ v16bf load_A_f32(const float* __restrict__ tile,
                                            int ld, int lane) {
  const int m = lane & 15, half = lane >> 4;
  const float* p = tile + (size_t)m * ld + half * 16;
  v16bf a;
#pragma unroll
  for (int v = 0; v < 8; ++v) {
    const int k0 = v * 2;
    a[2 * v]     = (__bf16)p[k0];
    a[2 * v + 1] = (__bf16)p[k0 + 1];
  }
  return a;
}

// Same as above but with per-lane row clamping (for partial node tiles).
__device__ __forceinline__ v16bf load_A_f32_clamp(const float* __restrict__ base,
                                                  int ld, int row0, int rowlim,
                                                  int kbase, int lane) {
  const int m = lane & 15, half = lane >> 4;
  int row = row0 + m;
  if (row > rowlim) row = rowlim;
  const float* p = base + (size_t)row * ld + kbase + half * 16;
  v16bf a;
#pragma unroll
  for (int v = 0; v < 8; ++v) {
    const int k0 = v * 2;
    a[2 * v]     = (__bf16)p[k0];
    a[2 * v + 1] = (__bf16)p[k0 + 1];
  }
  return a;
}

// Load 32x16 B tile from a pre-transposed bf16 weight WT[N][K]:
// per lane a single 32-byte contiguous read (two global_load_b128).
__device__ __forceinline__ v16bf load_B_T(const __bf16* __restrict__ Wt,
                                          int ldk, int ncol, int kbase, int lane) {
  const int n = ncol + (lane & 15), half = lane >> 4;
  const __bf16* p = Wt + (size_t)n * ldk + kbase + half * 16;
  v8u raw = *reinterpret_cast<const v8u*>(p);  // 32B aligned by construction
  return __builtin_bit_cast(v16bf, raw);
}

__device__ __forceinline__ int lower_bound_dev(const int* __restrict__ arr,
                                               int n, int key) {
  int lo = 0, hi = n;
  while (lo < hi) {
    int mid = (lo + hi) >> 1;
    if (arr[mid] < key) lo = mid + 1; else hi = mid;
  }
  return lo;
}

// ---------------------------------------------------------------------------
// Prep: transpose+convert a row-major f32 weight [K][N] into bf16 WT[N][K].
// kshift = log2(K). Grid covers N*K elements, K fastest (coalesced writes).
// ---------------------------------------------------------------------------
__global__ __launch_bounds__(256) void prep_weight_kernel(
    const float* __restrict__ W, __bf16* __restrict__ Wt, int N, int kshift) {
  const int i = blockIdx.x * 256 + threadIdx.x;
  const int K = 1 << kshift;
  const int n = i >> kshift, k = i & (K - 1);
  if (n < N) Wt[i] = (__bf16)W[(size_t)k * N + n];
}

// ---------------------------------------------------------------------------
// Kernel 1: Q = z_prev @ Wq + bq   [4096 x 256]
// Grid: B/16 blocks, 256 threads (8 waves, 32 cols each).
// ---------------------------------------------------------------------------
__global__ __launch_bounds__(256) void q_proj_kernel(
    const float* __restrict__ zprev, const __bf16* __restrict__ WqT,
    const float* __restrict__ bq, float* __restrict__ Qout) {
  const int tid = threadIdx.x, lane = tid & 31, wv = tid >> 5;
  const int r0 = blockIdx.x * 16;
  const int colbase = wv * 32;
  v8f acc0 = {}, acc1 = {};
#pragma unroll
  for (int kk = 0; kk < 8; ++kk) {
    v16bf a  = load_A_f32(zprev + (size_t)r0 * D_DIM + kk * 32, D_DIM, lane);
    v16bf b0 = load_B_T(WqT, D_DIM, colbase,      kk * 32, lane);
    v16bf b1 = load_B_T(WqT, D_DIM, colbase + 16, kk * 32, lane);
    acc0 = wmma_bf16(a, b0, acc0);
    acc1 = wmma_bf16(a, b1, acc1);
  }
  const int half = lane >> 4, nl = lane & 15;
#pragma unroll
  for (int r = 0; r < 8; ++r) {
    const int m = r + 8 * half;
    const int n0 = colbase + nl;
    Qout[(size_t)(r0 + m) * D_DIM + n0]      = acc0[r] + bq[n0];
    Qout[(size_t)(r0 + m) * D_DIM + n0 + 16] = acc1[r] + bq[n0 + 16];
  }
}

// ---------------------------------------------------------------------------
// Kernel 2: K-tile GEMM (kept in LDS only) + attention scores.
// scores[c,h] = dot(Q[batch[c], h*64:], K[c, h*64:]) / 8
// Grid: C/16 blocks, 256 threads.
// ---------------------------------------------------------------------------
__global__ __launch_bounds__(256) void kv_scores_kernel(
    const float* __restrict__ nodes, const __bf16* __restrict__ WkT,
    const float* __restrict__ bk, const float* __restrict__ Qws,
    const int* __restrict__ batch, float* __restrict__ scores_out) {
  __shared__ float Klds[16 * 264];
  __shared__ float red[256];
  const int tid = threadIdx.x, lane = tid & 31, wv = tid >> 5;
  const int c0 = blockIdx.x * 16;
  const int colbase = wv * 32;
  v8f acc0 = {}, acc1 = {};
#pragma unroll
  for (int kk = 0; kk < 8; ++kk) {
    v16bf a  = load_A_f32(nodes + (size_t)c0 * D_DIM + kk * 32, D_DIM, lane);
    v16bf b0 = load_B_T(WkT, D_DIM, colbase,      kk * 32, lane);
    v16bf b1 = load_B_T(WkT, D_DIM, colbase + 16, kk * 32, lane);
    acc0 = wmma_bf16(a, b0, acc0);
    acc1 = wmma_bf16(a, b1, acc1);
  }
  const int half = lane >> 4, nl = lane & 15;
#pragma unroll
  for (int r = 0; r < 8; ++r) {
    const int m = r + 8 * half;
    const int n0 = colbase + nl;
    Klds[m * 264 + n0]      = acc0[r] + bk[n0];
    Klds[m * 264 + n0 + 16] = acc1[r] + bk[n0 + 16];
  }
  __syncthreads();
  // 256 threads: (m,h) = tid>>2, part = tid&3 -> 16-wide partial dot each.
  {
    const int part = tid & 3;
    const int sid  = tid >> 2;           // 0..63
    const int m = sid >> 2, h = sid & 3;
    const int b = batch[c0 + m];
    const float* q    = Qws + (size_t)b * D_DIM + h * HD_DIM + part * 16;
    const float* krow = Klds + m * 264 + h * HD_DIM + part * 16;
    float s = 0.f;
#pragma unroll
    for (int hd = 0; hd < 16; ++hd) s += q[hd] * krow[hd];
    red[tid] = s;
  }
  __syncthreads();
  if (tid < 64) {
    const int m = tid >> 2, h = tid & 3;
    const float s = red[tid * 4] + red[tid * 4 + 1] + red[tid * 4 + 2] + red[tid * 4 + 3];
    scores_out[(c0 + m) * H_HEADS + h] = s * 0.125f;  // 1/sqrt(64)
  }
}

// ---------------------------------------------------------------------------
// Kernel 3: per-graph segment softmax + V recompute (WMMA) + weighted pool.
// One workgroup per graph; batch is sorted so nodes are contiguous.
// Grid: B blocks, 256 threads (8 waves, 32 cols each).
// ---------------------------------------------------------------------------
__global__ __launch_bounds__(256) void attn_pool_kernel(
    const float* __restrict__ nodes, const __bf16* __restrict__ WvT,
    const float* __restrict__ bv, const int* __restrict__ batch,
    const float* __restrict__ scores, float* __restrict__ pooled_out) {
  __shared__ int   se[2];
  __shared__ float red[256];
  __shared__ float smax_l[4];
  __shared__ float inv_l[4];
  __shared__ float attn_l[64];     // [m][h]
  __shared__ float pooled_l[256];
  const int tid = threadIdx.x, lane = tid & 31, wv = tid >> 5;
  const int g = blockIdx.x;

  if (tid == 0) se[0] = lower_bound_dev(batch, C_NODES, g);
  if (tid == 1) se[1] = lower_bound_dev(batch, C_NODES, g + 1);
  pooled_l[tid] = 0.f;
  __syncthreads();
  const int s = se[0], e = se[1];

  // --- segment max per head ---
  const int h_r = tid & 3, idx_r = tid >> 2;
  float lmax = -INFINITY;
  for (int c = s + idx_r; c < e; c += 64)
    lmax = fmaxf(lmax, scores[c * H_HEADS + h_r]);
  red[tid] = lmax;
  __syncthreads();
  if (tid < 4) {
    float m = red[tid];
    for (int i = 1; i < 64; ++i) m = fmaxf(m, red[tid + 4 * i]);
    smax_l[tid] = m;
  }
  __syncthreads();
  // --- segment exp-sum per head ---
  float lsum = 0.f;
  const float mh = smax_l[h_r];
  for (int c = s + idx_r; c < e; c += 64)
    lsum += expf(scores[c * H_HEADS + h_r] - mh);
  red[tid] = lsum;
  __syncthreads();
  if (tid < 4) {
    float d = 0.f;
    for (int i = 0; i < 64; ++i) d += red[tid + 4 * i];
    inv_l[tid] = 1.0f / (d + 1e-16f);
  }
  __syncthreads();

  // --- V GEMM per 16-node tile + weighted accumulate ---
  const int ntiles = (e - s + 15) >> 4;
  const int colbase = wv * 32;
  const int half = lane >> 4, nl = lane & 15;
  for (int t = 0; t < ntiles; ++t) {
    const int c0 = s + t * 16;
    __syncthreads();  // attn_l reuse guard
    if (tid < 64) {
      const int m = tid >> 2, hh = tid & 3;
      const int c = c0 + m;
      float w = 0.f;
      if (c < e) w = expf(scores[c * H_HEADS + hh] - smax_l[hh]) * inv_l[hh];
      attn_l[m * 4 + hh] = w;
    }
    __syncthreads();

    v8f acc0 = {}, acc1 = {};
#pragma unroll
    for (int kk = 0; kk < 8; ++kk) {
      v16bf a  = load_A_f32_clamp(nodes, D_DIM, c0, C_NODES - 1, kk * 32, lane);
      v16bf b0 = load_B_T(WvT, D_DIM, colbase,      kk * 32, lane);
      v16bf b1 = load_B_T(WvT, D_DIM, colbase + 16, kk * 32, lane);
      acc0 = wmma_bf16(a, b0, acc0);
      acc1 = wmma_bf16(a, b1, acc1);
    }
    // Contract with attention weights: pooled[n] += sum_m attn[m,h(n)] * (V+bv).
#pragma unroll
    for (int sub = 0; sub < 2; ++sub) {
      const int n = colbase + sub * 16 + nl;
      const int hh = n >> 6;
      const float bvn = bv[n];
      float partial = 0.f;
#pragma unroll
      for (int r = 0; r < 8; ++r) {
        const int m = r + 8 * half;
        const float vval = (sub == 0 ? acc0[r] : acc1[r]) + bvn;
        partial += attn_l[m * 4 + hh] * vval;
      }
      const float tot = partial + __shfl_xor(partial, 16);
      if (lane < 16) pooled_l[n] += tot;  // wave-private columns
    }
  }
  __syncthreads();
  pooled_out[(size_t)g * D_DIM + tid] = pooled_l[tid];
}

// ---------------------------------------------------------------------------
// Kernel 4: fused MLP (concat -> GEMM 512x256 -> exact GELU -> GEMM 256x256)
// + RMSNorm, all via WMMA with LDS staging.
// Grid: B/16 blocks, 256 threads.
// ---------------------------------------------------------------------------
__global__ __launch_bounds__(256) void mlp_rms_kernel(
    const float* __restrict__ pooled, const float* __restrict__ zprev,
    const __bf16* __restrict__ W1T, const float* __restrict__ b1,
    const __bf16* __restrict__ W2T, const float* __restrict__ b2,
    const float* __restrict__ scale, float* __restrict__ out) {
  __shared__ float z_lds[16 * 512];   // also reused for y
  __shared__ float h_lds[16 * 256];
  __shared__ float rms_l[16];
  const int tid = threadIdx.x, lane = tid & 31, wv = tid >> 5;
  const int r0 = blockIdx.x * 16;
  const int colbase = wv * 32;
  const int half = lane >> 4, nl = lane & 15;

  // stage z = concat(pooled, z_prev)
  for (int i = tid; i < 16 * 512; i += 256) {
    const int m = i >> 9, k = i & 511;
    z_lds[i] = (k < D_DIM) ? pooled[(size_t)(r0 + m) * D_DIM + k]
                           : zprev[(size_t)(r0 + m) * D_DIM + (k - D_DIM)];
  }
  __syncthreads();

  // h = gelu(z @ W1 + b1), K = 512
  v8f acc0 = {}, acc1 = {};
#pragma unroll
  for (int kk = 0; kk < 16; ++kk) {
    v16bf a   = load_A_f32(z_lds + kk * 32, 512, lane);
    v16bf bt0 = load_B_T(W1T, 2 * D_DIM, colbase,      kk * 32, lane);
    v16bf bt1 = load_B_T(W1T, 2 * D_DIM, colbase + 16, kk * 32, lane);
    acc0 = wmma_bf16(a, bt0, acc0);
    acc1 = wmma_bf16(a, bt1, acc1);
  }
#pragma unroll
  for (int r = 0; r < 8; ++r) {
    const int m = r + 8 * half;
#pragma unroll
    for (int sub = 0; sub < 2; ++sub) {
      const int n = colbase + sub * 16 + nl;
      const float x = (sub == 0 ? acc0[r] : acc1[r]) + b1[n];
      h_lds[m * 256 + n] = 0.5f * x * (1.0f + erff(x * 0.70710678118654752f));
    }
  }
  __syncthreads();

  // y = h @ W2 + b2, K = 256
  v8f acc2 = {}, acc3 = {};
#pragma unroll
  for (int kk = 0; kk < 8; ++kk) {
    v16bf a   = load_A_f32(h_lds + kk * 32, 256, lane);
    v16bf bt0 = load_B_T(W2T, D_DIM, colbase,      kk * 32, lane);
    v16bf bt1 = load_B_T(W2T, D_DIM, colbase + 16, kk * 32, lane);
    acc2 = wmma_bf16(a, bt0, acc2);
    acc3 = wmma_bf16(a, bt1, acc3);
  }
  float* y_lds = z_lds;  // reuse
#pragma unroll
  for (int r = 0; r < 8; ++r) {
    const int m = r + 8 * half;
#pragma unroll
    for (int sub = 0; sub < 2; ++sub) {
      const int n = colbase + sub * 16 + nl;
      y_lds[m * 256 + n] = (sub == 0 ? acc2[r] : acc3[r]) + b2[n];
    }
  }
  __syncthreads();

  // RMSNorm per row
  if (tid < 16) {
    float ss = 0.f;
    for (int n = 0; n < 256; ++n) {
      const float yv = y_lds[tid * 256 + n];
      ss += yv * yv;
    }
    rms_l[tid] = rsqrtf(ss * (1.0f / 256.0f) + 1e-6f);
  }
  __syncthreads();
  for (int m = 0; m < 16; ++m)
    out[(size_t)(r0 + m) * D_DIM + tid] = scale[tid] * y_lds[m * 256 + tid] * rms_l[m];
}

// ---------------------------------------------------------------------------
extern "C" void kernel_launch(void* const* d_in, const int* in_sizes, int n_in,
                              void* d_out, int out_size, void* d_ws, size_t ws_size,
                              hipStream_t stream) {
  const float* nodes = (const float*)d_in[0];
  const float* zprev = (const float*)d_in[1];
  const int*   batch = (const int*)d_in[2];
  // d_in[3] = num_graphs (fixed to B_GRAPH)
  const float* Wk = (const float*)d_in[4];
  const float* bk = (const float*)d_in[5];
  const float* Wv = (const float*)d_in[6];
  const float* bv = (const float*)d_in[7];
  const float* Wq = (const float*)d_in[8];
  const float* bq = (const float*)d_in[9];
  const float* W1 = (const float*)d_in[10];
  const float* b1 = (const float*)d_in[11];
  const float* W2 = (const float*)d_in[12];
  const float* b2 = (const float*)d_in[13];
  const float* scale = (const float*)d_in[14];
  float* out = (float*)d_out;

  // Workspace layout (all offsets keep >=32B alignment).
  float* Qws    = (float*)d_ws;                          // B*D   = 4 MB
  float* scores = Qws + (size_t)B_GRAPH * D_DIM;         // C*H   = 4 MB
  float* pooled = scores + (size_t)C_NODES * H_HEADS;    // B*D   = 4 MB
  __bf16* WkT = (__bf16*)(pooled + (size_t)B_GRAPH * D_DIM);
  __bf16* WvT = WkT + (size_t)D_DIM * D_DIM;             // 128 KB each
  __bf16* WqT = WvT + (size_t)D_DIM * D_DIM;
  __bf16* W1T = WqT + (size_t)D_DIM * D_DIM;             // 256x512 = 256 KB
  __bf16* W2T = W1T + (size_t)D_DIM * 2 * D_DIM;

  // One-time (per launch) weight transpose+convert to bf16 [N][K].
  prep_weight_kernel<<<(D_DIM * D_DIM) / 256, 256, 0, stream>>>(Wk, WkT, D_DIM, 8);
  prep_weight_kernel<<<(D_DIM * D_DIM) / 256, 256, 0, stream>>>(Wv, WvT, D_DIM, 8);
  prep_weight_kernel<<<(D_DIM * D_DIM) / 256, 256, 0, stream>>>(Wq, WqT, D_DIM, 8);
  prep_weight_kernel<<<(D_DIM * 2 * D_DIM) / 256, 256, 0, stream>>>(W1, W1T, D_DIM, 9);
  prep_weight_kernel<<<(D_DIM * D_DIM) / 256, 256, 0, stream>>>(W2, W2T, D_DIM, 8);

  q_proj_kernel<<<B_GRAPH / 16, 256, 0, stream>>>(zprev, WqT, bq, Qws);
  kv_scores_kernel<<<C_NODES / 16, 256, 0, stream>>>(nodes, WkT, bk, Qws, batch, scores);
  attn_pool_kernel<<<B_GRAPH, 256, 0, stream>>>(nodes, WvT, bv, batch, scores, pooled);
  mlp_rms_kernel<<<B_GRAPH / 16, 256, 0, stream>>>(pooled, zprev, W1T, b1, W2T, b2, scale, out);
}
